// Seq2Seq_10213432230508
// MI455X (gfx1250) — matrix operand order repaired and checked by reading, not compiled, not run on hardware
//
#include <hip/hip_runtime.h>

typedef __attribute__((ext_vector_type(16))) _Float16 v16h;
typedef __attribute__((ext_vector_type(8)))  _Float16 v8h;
typedef __attribute__((ext_vector_type(8)))  float    v8f;

constexpr int Bc = 32, Tc = 128, Ec = 128, Hc = 128, HHc = 64, Vc = 36294;

// ---------------------------------------------------------------------------
// WMMA helpers (CDNA5 v_wmma_f32_16x16x32_f16, wave32)
// ---------------------------------------------------------------------------
__device__ __forceinline__ v8f wmma16(v16h a, v16h b, v8f c) {
  // (neg_a, A, neg_b, B, c_mod, C, reuse_a, reuse_b)
  return __builtin_amdgcn_wmma_f32_16x16x32_f16(false, a, false, b, (short)0, c,
                                                false, false);
}

// A-fragment (16x32 f16, M x K) from row-major f16 (global or LDS).
// Lane l: row m0+(l&15); halves e: k = k0 + (l>>4)*8 + (e<8 ? e : e+8).
__device__ __forceinline__ v16h load_a_h16(const _Float16* A, int lda, int m0,
                                           int k0, int lane) {
  const _Float16* p = A + (size_t)(m0 + (lane & 15)) * lda + k0 + ((lane >> 4) << 3);
  v8h lo = *(const v8h*)p;
  v8h hi = *(const v8h*)(p + 16);
  v16h a;
#pragma unroll
  for (int e = 0; e < 8; ++e) { a[e] = lo[e]; a[8 + e] = hi[e]; }
  return a;
}

// Same fragment but source is fp32 (LDS state) — convert on the fly.
__device__ __forceinline__ v16h load_a_f32(const float* A, int lda, int m0,
                                           int k0, int lane) {
  const float* p = A + (size_t)(m0 + (lane & 15)) * lda + k0 + ((lane >> 4) << 3);
  v16h a;
#pragma unroll
  for (int e = 0; e < 8; ++e) { a[e] = (_Float16)p[e]; a[8 + e] = (_Float16)p[16 + e]; }
  return a;
}

// A-fragment over the virtual concatenation [h | ctx] (each 32x128 fp32 in LDS).
__device__ __forceinline__ v16h load_a_cat(const float* hbuf, const float* cbuf,
                                           int m0, int k0, int lane) {
  int m = m0 + (lane & 15);
  int kb = k0 + ((lane >> 4) << 3);
  v16h a;
#pragma unroll
  for (int e = 0; e < 16; ++e) {
    int k = kb + (e < 8 ? e : e + 8);
    float v = (k < 128) ? hbuf[m * 128 + k] : cbuf[m * 128 + (k - 128)];
    a[e] = (_Float16)v;
  }
  return a;
}

// B-fragment from pre-packed fragment-linear layout: fully coalesced 32B/lane.
__device__ __forceinline__ v16h load_b_pk(const _Float16* P, int nt, int kt,
                                          int KT, int lane) {
  return *(const v16h*)(P + ((((size_t)nt * KT + kt) * 32 + lane) << 4));
}

// ---------------------------------------------------------------------------
// Prep kernels
// ---------------------------------------------------------------------------
__global__ void k_lengths(const int* __restrict__ tokens, int* __restrict__ len) {
  int b = threadIdx.x;
  if (b >= Bc) return;
  int c = 0;
  for (int t = 0; t < Tc; ++t) c += (tokens[b * Tc + t] != 0);
  len[b] = c;
}

__global__ void k_embed(const int* __restrict__ tokens, const int* __restrict__ teacher,
                        const float* __restrict__ emb, _Float16* __restrict__ x_h,
                        _Float16* __restrict__ dx_h) {
  int t = blockIdx.x / Bc, b = blockIdx.x % Bc;
  int e = threadIdx.x;
  int tok = tokens[b * Tc + t];
  x_h[(size_t)blockIdx.x * Ec + e] = (_Float16)emb[(size_t)tok * Ec + e];
  int dtok = (t == 0) ? 1 : teacher[b * Tc + (t - 1)];  // SOS=1, teacher forcing
  dx_h[(size_t)blockIdx.x * Ec + e] = (_Float16)emb[(size_t)dtok * Ec + e];
}

__global__ void k_zero(float* __restrict__ p, int n) {
  int i = blockIdx.x * blockDim.x + threadIdx.x;
  if (i < n) p[i] = 0.0f;
}

// Pack W (O x K fp32, row-major; B = W^T) into WMMA B-fragment order.
// Lane l of tile (nt,kt): col n = nt*16 + (l&15); k = kt*32 + (l>>4)*8 + (e<8 ? e : e+8).
__global__ void k_pack(const float* __restrict__ W, _Float16* __restrict__ P,
                       int O, int K) {
  int KT = K >> 5, NT = (O + 15) >> 4;
  int idx = blockIdx.x * blockDim.x + threadIdx.x;
  if (idx >= NT * KT * 32) return;
  int lane = idx & 31;
  int kt = (idx >> 5) % KT;
  int nt = (idx >> 5) / KT;
  int n = nt * 16 + (lane & 15);
  int kbase = kt * 32 + ((lane >> 4) << 3);
  _Float16* out = P + ((size_t)idx << 4);
#pragma unroll
  for (int e = 0; e < 16; ++e) {
    float v = (n < O) ? W[(size_t)n * K + kbase + (e < 8 ? e : e + 8)] : 0.0f;
    out[e] = (_Float16)v;
  }
}

// ---------------------------------------------------------------------------
// Generic WMMA GEMM: C(MxO fp32, row-major) = A(MxK f16) @ packedB + bias
// One 16x16 C tile per wave; K-loop of 16x16x32 WMMA.
// ---------------------------------------------------------------------------
__global__ void k_gemm(const _Float16* __restrict__ A, const _Float16* __restrict__ P,
                       const float* __restrict__ bias, float* __restrict__ C,
                       int M, int K, int O) {
  int KT = K >> 5, NT = (O + 15) >> 4, MT = M >> 4;
  int wid = (blockIdx.x * blockDim.x + threadIdx.x) >> 5;
  int lane = threadIdx.x & 31;
  if (wid >= MT * NT) return;
  int nt = wid % NT, mt = wid / NT;
  v8f acc = {};
  for (int kt = 0; kt < KT; ++kt)
    acc = wmma16(load_a_h16(A, K, mt * 16, kt * 32, lane), load_b_pk(P, nt, kt, KT, lane), acc);
  int n = nt * 16 + (lane & 15);
  if (n >= O) return;
  float bb = bias ? bias[n] : 0.0f;
  int m0 = mt * 16 + ((lane >> 4) << 3);
#pragma unroll
  for (int v = 0; v < 8; ++v) C[(size_t)(m0 + v) * O + n] = acc[v] + bb;
}

// ---------------------------------------------------------------------------
// Encoder bi-GRU #2 recurrence. 2 blocks (dir 0 fwd / 1 bwd), 256 threads.
// Per step: gh = h @ whh^T + bhh via WMMA (2x12 tiles, K=64), then gates.
// Backward reversal-within-length folded into the time index.
// ---------------------------------------------------------------------------
__global__ void k_enc(const float* __restrict__ gxf, const float* __restrict__ gxb,
                      const _Float16* __restrict__ Pf, const _Float16* __restrict__ Pb,
                      const float* __restrict__ bhhf, const float* __restrict__ bhhb,
                      const int* __restrict__ len, float* __restrict__ enc,
                      float* __restrict__ h2) {
  const int dir = blockIdx.x;
  const float* gx = dir ? gxb : gxf;
  const _Float16* P = dir ? Pb : Pf;
  const float* bhh = dir ? bhhb : bhhf;
  __shared__ float h[Bc * HHc];      // 8 KB
  __shared__ float gh[Bc * 192];     // 24 KB
  int tid = threadIdx.x, lane = tid & 31, w = tid >> 5;  // 8 waves
  for (int i = tid; i < Bc * HHc; i += 256) h[i] = 0.0f;
  __syncthreads();
  for (int s = 0; s < Tc; ++s) {
    // gh = h @ whh^T + bhh : MT=2, NT=12, KT=2
    for (int tile = w; tile < 24; tile += 8) {
      int mt = tile / 12, nt = tile % 12;
      v8f acc = {};
#pragma unroll
      for (int kt = 0; kt < 2; ++kt)
        acc = wmma16(load_a_f32(h, HHc, mt * 16, kt * 32, lane), load_b_pk(P, nt, kt, 2, lane), acc);
      int n = nt * 16 + (lane & 15);
      int m0 = mt * 16 + ((lane >> 4) << 3);
#pragma unroll
      for (int v = 0; v < 8; ++v) gh[(m0 + v) * 192 + n] = acc[v] + bhh[n];
    }
    __syncthreads();
    // gates (r,z,n) + masked update + masked output write
    for (int i = tid; i < Bc * HHc; i += 256) {
      int b = i >> 6, j = i & 63;
      int lb = len[b];
      if (s < lb) {
        int trow = dir ? (lb - 1 - s) : s;
        const float* g = gx + ((size_t)trow * Bc + b) * 192;
        float r = 1.0f / (1.0f + __expf(-(g[j] + gh[b * 192 + j])));
        float z = 1.0f / (1.0f + __expf(-(g[64 + j] + gh[b * 192 + 64 + j])));
        float nn = tanhf(g[128 + j] + r * gh[b * 192 + 128 + j]);
        float hn = (1.0f - z) * nn + z * h[i];
        h[i] = hn;
        enc[((size_t)trow * Bc + b) * Hc + dir * HHc + j] = hn;
      }
    }
    __syncthreads();
  }
  for (int i = tid; i < Bc * HHc; i += 256) {
    int b = i >> 6, j = i & 63;
    h2[b * Hc + dir * HHc + j] = h[i];
  }
}

// ---------------------------------------------------------------------------
// Decoder: single resident workgroup, 1024 threads (32 waves), state in LDS.
// Per step: gh (WMMA 2x24, K=128) -> GRU -> dot-attn -> softmax -> ctx ->
// co = tanh([h,ctx]@w1^T + b1) (WMMA 2x8, K=256) -> co_h (f16, (b,t) rows).
// ---------------------------------------------------------------------------
__global__ void k_dec(const float* __restrict__ gx, const _Float16* __restrict__ Pwhh,
                      const float* __restrict__ dbhh, const float* __restrict__ h2,
                      const float* __restrict__ enc, const _Float16* __restrict__ Pw1,
                      const float* __restrict__ b1, _Float16* __restrict__ co_h) {
  __shared__ float smem[16384];              // 64 KB total
  float* h = smem;                           // 32x128, persistent
  float* gh = smem + 4096;                   // 32x384 (transient)
  float* att = smem + 4096;                  // 32x128, reuses gh region
  float* ctx = smem + 8192;                  // 32x128, reuses gh region
  int tid = threadIdx.x, lane = tid & 31, w = tid >> 5;  // 32 waves
  for (int i = tid; i < 4096; i += 1024) h[i] = h2[i];
  __syncthreads();
  for (int t = 0; t < Tc; ++t) {
    // gh = h @ dwhh^T + dbhh : MT=2, NT=24, KT=4 (192 wmma/step)
    for (int tile = w; tile < 48; tile += 32) {
      int mt = tile / 24, nt = tile % 24;
      v8f acc = {};
#pragma unroll
      for (int kt = 0; kt < 4; ++kt)
        acc = wmma16(load_a_f32(h, Hc, mt * 16, kt * 32, lane), load_b_pk(Pwhh, nt, kt, 4, lane), acc);
      int n = nt * 16 + (lane & 15);
      int m0 = mt * 16 + ((lane >> 4) << 3);
#pragma unroll
      for (int v = 0; v < 8; ++v) gh[(m0 + v) * 384 + n] = acc[v] + dbhh[n];
    }
    __syncthreads();
    // GRU cell
    const float* g = gx + (size_t)t * Bc * 384;
    for (int i = tid; i < 4096; i += 1024) {
      int b = i >> 7, j = i & 127;
      float r = 1.0f / (1.0f + __expf(-(g[b * 384 + j] + gh[b * 384 + j])));
      float z = 1.0f / (1.0f + __expf(-(g[b * 384 + 128 + j] + gh[b * 384 + 128 + j])));
      float nn = tanhf(g[b * 384 + 256 + j] + r * gh[b * 384 + 256 + j]);
      h[i] = (1.0f - z) * nn + z * h[i];
    }
    __syncthreads();
    // Luong dot scores: att[b,tp] = h[b]·enc[tp,b]   (gh region now free)
    for (int i = tid; i < 4096; i += 1024) {
      int b = i >> 7, tp = i & 127;
      const float* e = enc + ((size_t)tp * Bc + b) * Hc;
      const float* hb = h + b * Hc;
      float s = 0.0f;
      for (int k = 0; k < Hc; ++k) s += hb[k] * e[k];
      att[b * Tc + tp] = s;
    }
    __syncthreads();
    // softmax over T per batch row
    if (tid < Bc) {
      float mx = -1e30f;
      for (int tp = 0; tp < Tc; ++tp) mx = fmaxf(mx, att[tid * Tc + tp]);
      float sum = 0.0f;
      for (int tp = 0; tp < Tc; ++tp) { float ev = __expf(att[tid * Tc + tp] - mx); att[tid * Tc + tp] = ev; sum += ev; }
      float inv = 1.0f / sum;
      for (int tp = 0; tp < Tc; ++tp) att[tid * Tc + tp] *= inv;
    }
    __syncthreads();
    // ctx[b,j] = sum_tp att[b,tp] * enc[tp,b,j]
    for (int i = tid; i < 4096; i += 1024) {
      int b = i >> 7, j = i & 127;
      float s = 0.0f;
      for (int tp = 0; tp < Tc; ++tp) s += att[b * Tc + tp] * enc[((size_t)tp * Bc + b) * Hc + j];
      ctx[i] = s;
    }
    __syncthreads();
    // co = tanh([h,ctx] @ w1^T + b1) : MT=2, NT=8, KT=8 (128 wmma/step)
    // Store rows as (b, t) so the logits GEMM writes d_out linearly.
    for (int tile = w; tile < 16; tile += 32) {
      int mt = tile / 8, nt = tile % 8;
      v8f acc = {};
#pragma unroll
      for (int kt = 0; kt < 8; ++kt)
        acc = wmma16(load_a_cat(h, ctx, mt * 16, kt * 32, lane), load_b_pk(Pw1, nt, kt, 8, lane), acc);
      int n = nt * 16 + (lane & 15);
      int m0 = mt * 16 + ((lane >> 4) << 3);
#pragma unroll
      for (int v = 0; v < 8; ++v)
        co_h[((size_t)(m0 + v) * Tc + t) * Hc + n] = (_Float16)tanhf(acc[v] + b1[n]);
    }
    __syncthreads();
  }
}

// ---------------------------------------------------------------------------
// Output projection: logits(4096 x 36294, K=128). 38 GFLOP of f16 WMMA.
// 4 N-tiles per wave reuse each A fragment 4x; co_h rows are (b,t) so the
// store index is exactly m*V + n -> 8 consecutive output rows per tile.
// ---------------------------------------------------------------------------
constexpr int NTW = 4;  // N-tiles per wave
__global__ void k_logits(const _Float16* __restrict__ A, const _Float16* __restrict__ P,
                         const float* __restrict__ b2, float* __restrict__ out, int NT) {
  int wid = (blockIdx.x * blockDim.x + threadIdx.x) >> 5;
  int lane = threadIdx.x & 31;
  int MT = (Tc * Bc) >> 4;
  int NW = (NT + NTW - 1) / NTW;
  if (wid >= MT * NW) return;
  int ng = wid % NW, mt = wid / NW;
  int nt0 = ng * NTW;
  v8f acc[NTW] = {};
#pragma unroll
  for (int kt = 0; kt < 4; ++kt) {
    v16h a = load_a_h16(A, Hc, mt * 16, kt * 32, lane);
#pragma unroll
    for (int q = 0; q < NTW; ++q) {
      if (nt0 + q < NT)  // wave-uniform guard
        acc[q] = wmma16(a, load_b_pk(P, nt0 + q, kt, 4, lane), acc[q]);
    }
  }
  int m0 = mt * 16 + ((lane >> 4) << 3);
#pragma unroll
  for (int q = 0; q < NTW; ++q) {
    int nt = nt0 + q;
    if (nt >= NT) break;
    int n = nt * 16 + (lane & 15);
    if (n >= Vc) continue;
    float bb = b2[n];
#pragma unroll
    for (int v = 0; v < 8; ++v)
      out[(size_t)(m0 + v) * Vc + n] = acc[q][v] + bb;  // row m = b*T + t
  }
}

// ---------------------------------------------------------------------------
extern "C" void kernel_launch(void* const* d_in, const int* in_sizes, int n_in,
                              void* d_out, int out_size, void* d_ws, size_t ws_size,
                              hipStream_t stream) {
  (void)in_sizes; (void)n_in; (void)out_size; (void)ws_size;
  const int*   tokens  = (const int*)d_in[0];
  const int*   teacher = (const int*)d_in[1];
  const float* emb     = (const float*)d_in[2];
  // g1f/g1b (indices 3..10) intentionally unused: their outputs are discarded.
  const float* g2f_wih = (const float*)d_in[11];
  const float* g2f_whh = (const float*)d_in[12];
  const float* g2f_bih = (const float*)d_in[13];
  const float* g2f_bhh = (const float*)d_in[14];
  const float* g2b_wih = (const float*)d_in[15];
  const float* g2b_whh = (const float*)d_in[16];
  const float* g2b_bih = (const float*)d_in[17];
  const float* g2b_bhh = (const float*)d_in[18];
  const float* dec_wih = (const float*)d_in[19];
  const float* dec_whh = (const float*)d_in[20];
  const float* dec_bih = (const float*)d_in[21];
  const float* dec_bhh = (const float*)d_in[22];
  const float* w1      = (const float*)d_in[23];
  const float* b1      = (const float*)d_in[24];
  const float* w2      = (const float*)d_in[25];
  const float* b2      = (const float*)d_in[26];
  float* out = (float*)d_out;

  char* base = (char*)d_ws;
  size_t off = 0;
  auto alloc = [&](size_t bytes) -> void* {
    off = (off + 255) & ~(size_t)255;
    void* p = base + off;
    off += bytes;
    return p;
  };
  _Float16* x_h  = (_Float16*)alloc((size_t)Tc * Bc * Ec * 2);
  _Float16* dx_h = (_Float16*)alloc((size_t)Tc * Bc * Ec * 2);
  float* gxf = (float*)alloc((size_t)Tc * Bc * 192 * 4);
  float* gxb = (float*)alloc((size_t)Tc * Bc * 192 * 4);
  float* gxd = (float*)alloc((size_t)Tc * Bc * 384 * 4);
  float* enc = (float*)alloc((size_t)Tc * Bc * Hc * 4);
  float* h2  = (float*)alloc((size_t)Bc * Hc * 4);
  _Float16* co_h = (_Float16*)alloc((size_t)Tc * Bc * Hc * 2);
  int* lengths = (int*)alloc(Bc * 4);

  auto packBytes = [](int O, int K) -> size_t {
    return (size_t)((O + 15) / 16) * (K / 32) * 32 * 16 * 2;
  };
  _Float16* p_wihf = (_Float16*)alloc(packBytes(192, 128));
  _Float16* p_wihb = (_Float16*)alloc(packBytes(192, 128));
  _Float16* p_whhf = (_Float16*)alloc(packBytes(192, 64));
  _Float16* p_whhb = (_Float16*)alloc(packBytes(192, 64));
  _Float16* p_dwih = (_Float16*)alloc(packBytes(384, 128));
  _Float16* p_dwhh = (_Float16*)alloc(packBytes(384, 128));
  _Float16* p_w1   = (_Float16*)alloc(packBytes(128, 256));
  _Float16* p_w2   = (_Float16*)alloc(packBytes(Vc, 128));

  k_lengths<<<1, 32, 0, stream>>>(tokens, lengths);
  k_embed<<<Tc * Bc, Ec, 0, stream>>>(tokens, teacher, emb, x_h, dx_h);
  { int n = Tc * Bc * Hc; k_zero<<<(n + 255) / 256, 256, 0, stream>>>(enc, n); }

  auto pack = [&](const float* W, _Float16* P, int O, int K) {
    int tot = ((O + 15) / 16) * (K / 32) * 32;
    k_pack<<<(tot + 255) / 256, 256, 0, stream>>>(W, P, O, K);
  };
  pack(g2f_wih, p_wihf, 192, 128);
  pack(g2b_wih, p_wihb, 192, 128);
  pack(g2f_whh, p_whhf, 192, 64);
  pack(g2b_whh, p_whhb, 192, 64);
  pack(dec_wih, p_dwih, 384, 128);
  pack(dec_whh, p_dwhh, 384, 128);
  pack(w1, p_w1, 128, 256);
  pack(w2, p_w2, Vc, 128);

  auto gemm = [&](const _Float16* A, const _Float16* P, const float* bias, float* C,
                  int M, int K, int O) {
    int waves = (M / 16) * ((O + 15) / 16);
    int thr = waves * 32;
    k_gemm<<<(thr + 255) / 256, 256, 0, stream>>>(A, P, bias, C, M, K, O);
  };
  gemm(x_h, p_wihf, g2f_bih, gxf, Tc * Bc, 128, 192);
  gemm(x_h, p_wihb, g2b_bih, gxb, Tc * Bc, 128, 192);
  gemm(dx_h, p_dwih, dec_bih, gxd, Tc * Bc, 128, 384);

  k_enc<<<2, 256, 0, stream>>>(gxf, gxb, p_whhf, p_whhb, g2f_bhh, g2b_bhh, lengths, enc, h2);
  k_dec<<<1, 1024, 0, stream>>>(gxd, p_dwhh, dec_bhh, h2, enc, p_w1, b1, co_h);

  int NTv = (Vc + 15) / 16;                       // 2269
  int NW = (NTv + NTW - 1) / NTW;                 // 568 tile-groups
  int waves = ((Tc * Bc) / 16) * NW;              // 145,408 waves
  int thr = waves * 32;
  k_logits<<<(thr + 255) / 256, 256, 0, stream>>>(co_h, p_w2, b2, out, NTv);
}
